// MultiHeadAttention_5540507812402
// MI455X (gfx1250) — compile-verified, matrix operand order checked by reading, not run on hardware
//
#include <hip/hip_runtime.h>
#include <hip/hip_bf16.h>
#include <math.h>

// ---------------------------------------------------------------------------
// MI455X (gfx1250, wave32) multi-head attention forward.
//   out = ((softmax_masked((XWq)(KWq)^T/8) (VWq)) concat-heads) @ Wo
// All matrix math via V_WMMA_F32_16X16X32_BF16 (f32 accumulate); softmax
// row-sums via a ones-matrix WMMA; row-max via v_permlane16 (VALU, no LDS).
// Workspace (needs >= 40 MB): Qp bf16 (8MB, pre-scaled by 1/8) | Kp bf16 (8MB)
// | Vt bf16 (8MB, transposed [D][B*S]) | Ao f32 (16MB).
// ---------------------------------------------------------------------------

typedef __attribute__((ext_vector_type(16))) __bf16 v16bf;
typedef __attribute__((ext_vector_type(8)))  __bf16 v8bf;
typedef __attribute__((ext_vector_type(8)))  float  v8f;

#define NEGV (-1.0e6f)
#define LSTR 40   // 32 + 8 pad; 80B row stride keeps 16B alignment, spreads banks

__device__ __forceinline__ v8f wmma_bf16(v16bf a, v16bf b, v8f c) {
    // (neg_a, A, neg_b, B, c_mod, C, reuse_a, reuse_b)
    return __builtin_amdgcn_wmma_f32_16x16x32_bf16(false, a, false, b,
                                                   (short)0, c, false, false);
}

// Cross-lane XOR-K exchange within each 16-lane half (K in {1,2,4,8}).
// Primary: v_permlane16_b32 (pure VALU, no dscnt). Fallback: ds_swizzle
// xor-mask mode (constant immediate, so both compile passes type-check).
template <int KX>
__device__ __forceinline__ float lanexor16(float x) {
    int xi = __float_as_int(x);
#if defined(__AMDGCN__) && __has_builtin(__builtin_amdgcn_permlane16)
    // nibble j of {hi,lo} = j ^ KX (source lane select within 16-lane row)
    constexpr unsigned lo = (KX == 1) ? 0x67452301u : (KX == 2) ? 0x54761032u
                          : (KX == 4) ? 0x32107654u : 0xFEDCBA98u;
    constexpr unsigned hi = (KX == 1) ? 0xEFCDAB89u : (KX == 2) ? 0xDCFE98BAu
                          : (KX == 4) ? 0xBA98FEDCu : 0x76543210u;
    int r = __builtin_amdgcn_permlane16(xi, xi, (int)lo, (int)hi, false, false);
#else
    int r = __builtin_amdgcn_ds_swizzle(xi, (KX << 10) | 0x1f);  // xor_mask=KX
#endif
    return __int_as_float(r);
}

// A/B fragment (16x32 bf16, contraction-contiguous row, LDS or global).
// Lane layout per ISA 7.12.2: lanes 0-15 hold K {0..7,16..23}, lanes 16-31
// hold K {8..15,24..31} of row/col (lane & 15).  Two 16B vector loads.
__device__ __forceinline__ v16bf load_frag(const __bf16* rowbase, int hi) {
    const v8bf a = *(const v8bf*)(rowbase + hi * 8);
    const v8bf b = *(const v8bf*)(rowbase + 16 + hi * 8);
    v16bf r;
#pragma unroll
    for (int i = 0; i < 8; ++i) { r[i] = a[i]; r[i + 8] = b[i]; }
    return r;
}

// ---------------------------------------------------------------------------
// GEMM: C[M,N] = cscale * A[M,K] * B[K,N], f32 inputs, bf16 WMMA internally.
// Block 256 threads = 8 waves; block tile 64x128, K-step 32.
// Wave (wm 0..3, wn 0..1): rows wm*16..+16, cols wn*64..+64 (4 WMMAs/A-frag).
// Epilogue modes: f32 row-major | bf16 row-major | bf16 transposed (C^T[N][M]).
// ---------------------------------------------------------------------------
enum { OUT_F32 = 0, OUT_BF16 = 1, OUT_BF16_T = 2 };

template <int MODE>
__global__ __launch_bounds__(256)
void gemm_bf16wmma(const float* __restrict__ A, const float* __restrict__ B,
                   void* __restrict__ Cout, int M, int N, int K, float cscale) {
    __shared__ alignas(16) __bf16 sA[64 * LSTR];    // [m][k]
    __shared__ alignas(16) __bf16 sB[128 * LSTR];   // [n][k] (transposed fill)

    const int tid  = threadIdx.x;
    const int lane = tid & 31;
    const int w    = tid >> 5;
    const int wm   = w & 3;
    const int wn   = w >> 2;
    const int hi   = lane >> 4;
    const int lm   = lane & 15;
    const int bm   = blockIdx.x * 64;
    const int bn   = blockIdx.y * 128;

    v8f acc[4] = { {}, {}, {}, {} };

    for (int k0 = 0; k0 < K; k0 += 32) {
        // --- A tile: 64x32 f32 -> bf16 LDS; 512 float4s over 256 threads.
#pragma unroll
        for (int i = 0; i < 2; ++i) {
            int idx  = tid * 2 + i;
            int row  = idx >> 3;            // 8 float4 per row
            int col4 = (idx & 7) * 4;
            const float4 v = *(const float4*)(A + (size_t)(bm + row) * K + k0 + col4);
            __bf16* dst = sA + row * LSTR + col4;
            dst[0] = (__bf16)v.x; dst[1] = (__bf16)v.y;
            dst[2] = (__bf16)v.z; dst[3] = (__bf16)v.w;
        }
        // --- B tile: 32x128 f32, stored transposed as sB[n][k].
        //     Unit = (k,k+1) x 4n: two b128 loads -> 4 packed ds_store_b32.
#pragma unroll
        for (int i = 0; i < 2; ++i) {
            int u  = tid * 2 + i;           // 512 units
            int kp = u & 15;                // k = 2*kp
            int nq = u >> 4;                // n = nq*4
            const float4 v0 = *(const float4*)(B + (size_t)(k0 + 2*kp    ) * N + bn + nq*4);
            const float4 v1 = *(const float4*)(B + (size_t)(k0 + 2*kp + 1) * N + bn + nq*4);
            const float a0[4] = { v0.x, v0.y, v0.z, v0.w };
            const float a1[4] = { v1.x, v1.y, v1.z, v1.w };
#pragma unroll
            for (int j = 0; j < 4; ++j) {
                union { __bf16 h[2]; unsigned u32; } p;
                p.h[0] = (__bf16)a0[j];     // k
                p.h[1] = (__bf16)a1[j];     // k+1
                *(unsigned*)(sB + (nq*4 + j) * LSTR + 2*kp) = p.u32;
            }
        }
        // prefetch next K tile (global_prefetch_b8)
        if (k0 + 32 < K) {
            __builtin_prefetch(A + (size_t)(bm + (tid >> 2)) * K + k0 + 32, 0, 0);
            __builtin_prefetch(B + (size_t)(k0 + 32 + (tid >> 3)) * N + bn + ((tid & 7) << 4), 0, 0);
        }
        __syncthreads();

        v16bf af = load_frag(sA + (wm * 16 + lm) * LSTR, hi);
#pragma unroll
        for (int g = 0; g < 4; ++g) {
            v16bf bfr = load_frag(sB + (wn * 64 + g * 16 + lm) * LSTR, hi);
            acc[g] = wmma_bf16(af, bfr, acc[g]);
        }
        __syncthreads();
    }

    // D layout: element r of lane -> row (r + 8*hi), col (lane&15) of the tile
    if (MODE == OUT_F32) {
        float* C = (float*)Cout;
#pragma unroll
        for (int g = 0; g < 4; ++g)
#pragma unroll
            for (int r = 0; r < 8; ++r) {
                int row = bm + wm * 16 + r + 8 * hi;
                C[(size_t)row * N + bn + wn * 64 + g * 16 + lm] = acc[g][r] * cscale;
            }
    } else if (MODE == OUT_BF16) {
        __bf16* C = (__bf16*)Cout;
#pragma unroll
        for (int g = 0; g < 4; ++g)
#pragma unroll
            for (int r = 0; r < 8; ++r) {
                int row = bm + wm * 16 + r + 8 * hi;
                C[(size_t)row * N + bn + wn * 64 + g * 16 + lm] = (__bf16)(acc[g][r] * cscale);
            }
    } else {  // OUT_BF16_T: C^T[N][M]; lane's 8 rows are contiguous -> b128 store
        __bf16* C = (__bf16*)Cout;
        const int m0 = bm + wm * 16 + 8 * hi;
#pragma unroll
        for (int g = 0; g < 4; ++g) {
            const int n = bn + wn * 64 + g * 16 + lm;
            v8bf pk;
#pragma unroll
            for (int r = 0; r < 8; ++r) pk[r] = (__bf16)(acc[g][r] * cscale);
            *(v8bf*)(C + (size_t)n * M + m0) = pk;
        }
    }
}

// ---------------------------------------------------------------------------
// Flash attention: one wave handles 16 query rows of one (b, h).
// Q (pre-scaled by 1/8) / K bf16 row-major [B*S][D]; V bf16 transposed
// [D][B*S] so every fragment is two contiguous 16B loads.
// Streaming masked softmax: row-max via permlane16, row-sum via ones-WMMA.
// ---------------------------------------------------------------------------
__global__ __launch_bounds__(32)
void flash_attn_kernel(const __bf16* __restrict__ Qp, const __bf16* __restrict__ Kp,
                       const __bf16* __restrict__ Vt, const int* __restrict__ vlen,
                       float* __restrict__ Op) {
    constexpr int S = 1024, D = 1024, HD = 64, M = 4096;
    __shared__ alignas(16) __bf16 sP[16 * LSTR];   // P tile 16x32 bf16 (+pad)

    const int lane = threadIdx.x & 31;
    const int hi   = lane >> 4;
    const int lm   = lane & 15;
    const int q0   = blockIdx.x * 16;
    const int b    = blockIdx.y >> 4;
    const int h    = blockIdx.y & 15;
    const int vl   = vlen[b];

    // ones B-matrix for row-sum WMMA
    v16bf ones;
#pragma unroll
    for (int i = 0; i < 16; ++i) ones[i] = (__bf16)1.0f;

    // Q fragments (rows q0..q0+15, hd split into two K-steps of 32)
    const __bf16* Qbase = Qp + ((size_t)(b * S + q0 + lm)) * D + h * HD;
    v16bf qf[2];
#pragma unroll
    for (int ks = 0; ks < 2; ++ks) qf[ks] = load_frag(Qbase + ks * 32, hi);

    float m[8], l[8];
    v8f o[4] = { {}, {}, {}, {} };
#pragma unroll
    for (int r = 0; r < 8; ++r) { m[r] = -3.0e38f; l[r] = 0.0f; }

    for (int k0 = 0; k0 < vl; k0 += 32) {       // vl is uniform per block
        // ---- scores: 16q x 32k = two 16x16 D frags, contraction hd=64 ----
        v8f s[2] = { {}, {} };
#pragma unroll
        for (int g = 0; g < 2; ++g) {
            const int key = k0 + g * 16 + lm;   // < S since S % 32 == 0
            const __bf16* Kbase = Kp + ((size_t)(b * S + key)) * D + h * HD;
#pragma unroll
            for (int ks = 0; ks < 2; ++ks) {
                v16bf kf = load_frag(Kbase + ks * 32, hi);
                s[g] = wmma_bf16(qf[ks], kf, s[g]);
            }
        }
        // ---- length mask (depends only on key column; scale is in Q) ----
#pragma unroll
        for (int g = 0; g < 2; ++g) {
            const bool valid = (k0 + g * 16 + lm) < vl;
#pragma unroll
            for (int r = 0; r < 8; ++r) s[g][r] = valid ? s[g][r] : NEGV;
        }
        // ---- row max across 16 lanes of each half: pure-VALU permlane ----
        float mx[8];
#pragma unroll
        for (int r = 0; r < 8; ++r) mx[r] = fmaxf(s[0][r], s[1][r]);
#pragma unroll
        for (int r = 0; r < 8; ++r) mx[r] = fmaxf(mx[r], lanexor16<1>(mx[r]));
#pragma unroll
        for (int r = 0; r < 8; ++r) mx[r] = fmaxf(mx[r], lanexor16<2>(mx[r]));
#pragma unroll
        for (int r = 0; r < 8; ++r) mx[r] = fmaxf(mx[r], lanexor16<4>(mx[r]));
#pragma unroll
        for (int r = 0; r < 8; ++r) mx[r] = fmaxf(mx[r], lanexor16<8>(mx[r]));

        float alpha[8];
#pragma unroll
        for (int r = 0; r < 8; ++r) {
            float mn = fmaxf(m[r], mx[r]);
            alpha[r] = __expf(m[r] - mn);
            m[r] = mn;
        }
        // ---- P = exp(s - m) ----
#pragma unroll
        for (int g = 0; g < 2; ++g)
#pragma unroll
            for (int r = 0; r < 8; ++r) s[g][r] = __expf(s[g][r] - m[r]);

        // ---- rearrange P (D layout) -> A layout through LDS ----
        __syncthreads();
#pragma unroll
        for (int g = 0; g < 2; ++g)
#pragma unroll
            for (int r = 0; r < 8; ++r)
                sP[(r + 8 * hi) * LSTR + g * 16 + lm] = (__bf16)s[g][r];
        __syncthreads();
        v16bf pf = load_frag(sP + lm * LSTR, hi);

        // ---- row sums on the matrix pipe: P x ones -> per-row sum in D layout
        v8f zero = {};
        v8f rsum = wmma_bf16(pf, ones, zero);
#pragma unroll
        for (int r = 0; r < 8; ++r) l[r] = l[r] * alpha[r] + rsum[r];
#pragma unroll
        for (int g2 = 0; g2 < 4; ++g2)
#pragma unroll
            for (int r = 0; r < 8; ++r) o[g2][r] *= alpha[r];

        // ---- O += P (16x32) x V (32xHD): V^T rows are key-contiguous ----
#pragma unroll
        for (int g2 = 0; g2 < 4; ++g2) {
            const __bf16* Vrow = Vt + (size_t)(h * HD + g2 * 16 + lm) * M + (b * S + k0);
            v16bf vf = load_frag(Vrow, hi);
            o[g2] = wmma_bf16(pf, vf, o[g2]);
        }
    }

    // ---- epilogue: normalize and store with head-concat layout ----
#pragma unroll
    for (int r = 0; r < 8; ++r) l[r] = 1.0f / l[r];
#pragma unroll
    for (int g2 = 0; g2 < 4; ++g2)
#pragma unroll
        for (int r = 0; r < 8; ++r) {
            int row = q0 + r + 8 * hi;
            Op[((size_t)(b * S + row)) * D + h * HD + g2 * 16 + lm] = o[g2][r] * l[r];
        }
}

// ---------------------------------------------------------------------------
extern "C" void kernel_launch(void* const* d_in, const int* in_sizes, int n_in,
                              void* d_out, int out_size, void* d_ws, size_t ws_size,
                              hipStream_t stream) {
    const float* queries = (const float*)d_in[0];
    const float* keys    = (const float*)d_in[1];
    const float* values  = (const float*)d_in[2];
    const int*   vlen    = (const int*)d_in[3];
    const float* W_q     = (const float*)d_in[4];
    const float* W_o     = (const float*)d_in[5];
    float* out = (float*)d_out;

    constexpr int Bb = 4, S = 1024, D = 1024;
    constexpr int M = Bb * S;                 // 4096
    const size_t matElems = (size_t)M * D;    // 4M elements

    char* ws = (char*)d_ws;                   // needs ws_size >= 40 MB
    __bf16* Qp = (__bf16*)ws;                                   // 8 MB
    __bf16* Kp = (__bf16*)(ws + matElems * sizeof(__bf16));     // 8 MB
    __bf16* Vt = (__bf16*)(ws + 2 * matElems * sizeof(__bf16)); // 8 MB
    float*  Ao = (float*) (ws + 3 * matElems * sizeof(__bf16)); // 16 MB

    dim3 gGemm(M / 64, D / 128);              // (64, 8)
    // Q projection pre-scaled by 1/sqrt(hd)=0.125 (exact power of two in bf16)
    gemm_bf16wmma<OUT_BF16  ><<<gGemm, 256, 0, stream>>>(queries, W_q, Qp, M, D, D, 0.125f);
    gemm_bf16wmma<OUT_BF16  ><<<gGemm, 256, 0, stream>>>(keys,    W_q, Kp, M, D, D, 1.0f);
    gemm_bf16wmma<OUT_BF16_T><<<gGemm, 256, 0, stream>>>(values,  W_q, Vt, M, D, D, 1.0f);

    dim3 gAttn(S / 16, Bb * 16);              // (64, 64) one wave each
    flash_attn_kernel<<<gAttn, 32, 0, stream>>>(Qp, Kp, Vt, vlen, Ao);

    gemm_bf16wmma<OUT_F32><<<gGemm, 256, 0, stream>>>(Ao, W_o, out, M, D, D, 1.0f);
}
